// P1GCN_6451040878746
// MI455X (gfx1250) — matrix operand order repaired and checked by reading, not compiled, not run on hardware
//
#include <hip/hip_runtime.h>
#include <hip/hip_bf16.h>

// ---------------------------------------------------------------------------
// P1GCN on MI455X (gfx1250): per layer
//   agg = segment_sum(h[src], dst)            (f32 gather + native f32 atomics)
//   out = relu_dropout(concat(h,agg) @ W + b)  (bf16 WMMA, f32 accumulate)
// Roofline: GEMM ~11.7 GFLOP (negligible at WMMA rates); traffic-dominated by
// edge gather + scatter-atomics (~1.8 GB -> ~80us at 23.3 TB/s, L2-resident).
// ---------------------------------------------------------------------------

typedef __attribute__((ext_vector_type(16))) __bf16 bf16x16;
typedef __attribute__((ext_vector_type(8)))  __bf16 bf16x8;
typedef __attribute__((ext_vector_type(8)))  float  f32x8;

__device__ __forceinline__ unsigned short f32_to_bf16_bits(float f) {
  unsigned u = __builtin_bit_cast(unsigned, f);
  unsigned r = u + 0x7FFFu + ((u >> 16) & 1u);   // round-to-nearest-even
  return (unsigned short)(r >> 16);
}

__device__ __forceinline__ float hash_uniform(unsigned x, unsigned seed) {
  x ^= seed * 0x9E3779B9u;
  x ^= x >> 16; x *= 0x7FEB352Du;
  x ^= x >> 15; x *= 0x846CA68Bu;
  x ^= x >> 16;
  return (float)(x >> 8) * (1.0f / 16777216.0f);
}

__device__ __forceinline__ bf16x16 load_frag(const unsigned short* p, int gap) {
  // 16 bf16 = two 16B chunks 'gap' elements apart (ds_load_b128/global_load_b128)
  bf16x8 lo = *(const bf16x8*)p;
  bf16x8 hi = *(const bf16x8*)(p + gap);
  bf16x16 r;
#pragma unroll
  for (int q = 0; q < 8; ++q) { r[q] = lo[q]; r[8 + q] = hi[q]; }
  return r;
}

// ---- zero fill (float4 stores; count is in float4 units) -------------------
__global__ void zero_f32x4(float* __restrict__ p, long long n4) {
  long long i = (long long)blockIdx.x * blockDim.x + threadIdx.x;
  if (i < n4) *(float4*)(p + i * 4) = make_float4(0.f, 0.f, 0.f, 0.f);
}

// ---- weight prep: W[K][Fo] f32 -> Wt[Fo_pad][K] bf16 (zero-padded cols) ----
__global__ void prep_wt(const float* __restrict__ W, unsigned short* __restrict__ Wt,
                        int K, int Fo, int Fo_pad) {
  int i = blockIdx.x * blockDim.x + threadIdx.x;
  if (i >= Fo_pad * K) return;
  int n = i / K, k = i - n * K;
  float v = (n < Fo) ? W[(size_t)k * Fo + n] : 0.0f;
  Wt[(size_t)n * K + k] = f32_to_bf16_bits(v);
}

// ---- edge scatter: agg[dst] += h[src], float4 chunks, native f32 atomics ---
__global__ void scatter_add_edges(const float* __restrict__ h,
                                  const long long* __restrict__ src,
                                  const long long* __restrict__ dst,
                                  float* __restrict__ agg,
                                  int E, int F) {
  const int chunks = F >> 2;
  long long i = (long long)blockIdx.x * blockDim.x + threadIdx.x;
  if (i >= (long long)E * chunks) return;
  int e = (int)(i / chunks);
  int c = (int)(i - (long long)e * chunks) << 2;
  long long s = src[e], d = dst[e];
  const float4 v = *(const float4*)(h + s * (long long)F + c);
  float* dp = agg + d * (long long)F + c;
  // unsafeAtomicAdd -> global_atomic_add_f32 (no CAS expansion) on AMD
  unsafeAtomicAdd(dp + 0, v.x);
  unsafeAtomicAdd(dp + 1, v.y);
  unsafeAtomicAdd(dp + 2, v.z);
  unsafeAtomicAdd(dp + 3, v.w);
}

// ---- fused WMMA GEMM: out = epilogue(concat(h,agg)[32 rows] @ Wt^T + b) ----
// Block = 256 threads = 8 waves; block owns 32 rows (2 m-subtiles of 16).
// Wave w: m-subtile = w&1, n-group = w>>1.  For Fo=128 each wave carries TWO
// accumulators sharing one A fragment per K-step (halves LDS traffic); all
// activity decisions are compile-time (no EXEC manipulation near WMMA).
template<int KD, int FO_PAD, int FO>
__global__ __launch_bounds__(256)
void gemm_wmma_fused(const float* __restrict__ h, const float* __restrict__ agg,
                     const unsigned short* __restrict__ Wt,
                     const float* __restrict__ bias, float* __restrict__ out,
                     int relu_drop, unsigned seed) {
  constexpr int F       = KD / 2;
  constexpr int TILES_N = FO_PAD / 16;
  constexpr int G       = (TILES_N >= 8) ? 2 : 1;   // n-tiles per wave
  constexpr int NGRPS   = (TILES_N + G - 1) / G;    // active n-groups: 4,4,1
  __shared__ alignas(16) unsigned short sA[32 * KD];   // <= 16 KB
  const int mblock = blockIdx.x * 32;

  // Stage A tile = [h | agg] rows, converted f32 -> bf16 (float4 granularity;
  // F is a multiple of 4 so chunks never straddle the h/agg boundary).
#pragma unroll
  for (int i = threadIdx.x * 4; i < 32 * KD; i += 256 * 4) {
    int r = i / KD;                    // compile-time shift (KD is constexpr)
    int k = i - r * KD;
    int row = mblock + r;
    const float* sp = (k < F) ? (h + (size_t)row * F + k)
                              : (agg + (size_t)row * F + (k - F));
    float4 v = *(const float4*)sp;
    unsigned short* q = sA + r * KD + k;
    q[0] = f32_to_bf16_bits(v.x);
    q[1] = f32_to_bf16_bits(v.y);
    q[2] = f32_to_bf16_bits(v.z);
    q[3] = f32_to_bf16_bits(v.w);
  }
  __syncthreads();

  const int wave = threadIdx.x >> 5;
  const int lane = threadIdx.x & 31;
  const int half = lane >> 4;          // 0: lanes 0-15, 1: lanes 16-31
  const int l15  = lane & 15;
  const int msub = wave & 1;
  const int ngrp = wave >> 1;          // 0..3

  // Only the Fo=8 layer leaves waves idle; retire them with one SCALAR branch
  // (readfirstlane -> s_cbranch) so EXEC is untouched in the WMMA region.
  if constexpr (NGRPS < 4) {
    if (__builtin_amdgcn_readfirstlane(ngrp) >= NGRPS) return;  // no later barrier
  }

  // A fragment: row = msub*16 + l15; lanes 0-15 carry K {0..7,16..23},
  // lanes 16-31 carry K {8..15,24..31}  (ISA 7.12.2, 16-bit A 16x32).
  const unsigned short* Ab = sA + (msub * 16 + l15) * KD + (half ? 8 : 0);

  // B fragments: column = ntile*16 + l15; lanes 0-15 carry K {0..7,16..23},
  // lanes 16-31 carry K {8..15,24..31} of each 32x16 K-step (matches A).
  const int nt0 = ngrp * G;
  const unsigned short* Bb0 = Wt + (size_t)(nt0 * 16 + l15) * KD + (half ? 8 : 0);
  const unsigned short* Bb1 = Bb0 + (size_t)16 * KD;   // used only when G==2

  f32x8 acc0 = {};
  f32x8 acc1 = {};
#pragma unroll
  for (int k0 = 0; k0 < KD; k0 += 32) {
    bf16x16 a  = load_frag(Ab + k0, 16);
    bf16x16 b0 = load_frag(Bb0 + k0, 16);
    acc0 = __builtin_amdgcn_wmma_f32_16x16x32_bf16(
        false, a, false, b0, (short)0, acc0, false, false);
    if constexpr (G == 2) {
      bf16x16 b1 = load_frag(Bb1 + k0, 16);
      acc1 = __builtin_amdgcn_wmma_f32_16x16x32_bf16(
          false, a, false, b1, (short)0, acc1, false, false);
    }
  }

  // Epilogue: bias + ReLU + deterministic hash dropout, f32 store.
  auto epilogue = [&](const f32x8& acc, int ntile) {
    const int ncol = ntile * 16 + l15;
    if (ncol < FO) {                   // divergent only for Fo=8 padding
      const float bv = bias[ncol];
#pragma unroll
      for (int v = 0; v < 8; ++v) {
        // C layout: VGPR v, lanes 0-15 -> M=v, lanes 16-31 -> M=v+8; N=l15.
        const int row = mblock + msub * 16 + v + half * 8;
        float val = acc[v] + bv;
        if (relu_drop) {
          val = fmaxf(val, 0.0f);
          float u = hash_uniform((unsigned)row * (unsigned)FO + (unsigned)ncol, seed);
          val = (u < 0.4f) ? val * 2.5f : 0.0f;  // keep-prob 0.4, scale 1/0.4
        }
        out[(size_t)row * FO + ncol] = val;
      }
    }
  };
  epilogue(acc0, nt0);
  if constexpr (G == 2) epilogue(acc1, nt0 + 1);
}

// ---------------------------------------------------------------------------
extern "C" void kernel_launch(void* const* d_in, const int* in_sizes, int n_in,
                              void* d_out, int out_size, void* d_ws, size_t ws_size,
                              hipStream_t stream) {
  (void)n_in; (void)out_size; (void)ws_size;
  const float*      x  = (const float*)d_in[0];
  const long long*  ei = (const long long*)d_in[1];   // int64 edge_index [2][E]
  const float* W1 = (const float*)d_in[2]; const float* b1 = (const float*)d_in[3];
  const float* W2 = (const float*)d_in[4]; const float* b2 = (const float*)d_in[5];
  const float* W3 = (const float*)d_in[6]; const float* b3 = (const float*)d_in[7];
  const float* W4 = (const float*)d_in[8]; const float* b4 = (const float*)d_in[9];

  const int N = in_sizes[0] / 128;     // 100000 (divisible by 32)
  const int E = in_sizes[1] / 2;       // 600000

  // Workspace carve-out (256B-aligned slices)
  char* ws = (char*)d_ws;
  size_t off = 0;
  auto carve = [&](size_t bytes) -> void* {
    void* p = ws + off;
    off = (off + bytes + 255) & ~(size_t)255;
    return p;
  };
  float* bufA = (float*)carve((size_t)N * 128 * sizeof(float));
  float* bufB = (float*)carve((size_t)N * 128 * sizeof(float));
  float* agg  = (float*)carve((size_t)N * 128 * sizeof(float));
  unsigned short* Wt1 = (unsigned short*)carve((size_t)128 * 256 * 2);
  unsigned short* Wt2 = (unsigned short*)carve((size_t) 64 * 256 * 2);
  unsigned short* Wt3 = (unsigned short*)carve((size_t) 64 * 128 * 2);
  unsigned short* Wt4 = (unsigned short*)carve((size_t) 16 * 128 * 2);

  auto cdiv = [](long long a, long long b) { return (unsigned)((a + b - 1) / b); };

  // Weight transpose + bf16 convert (tiny; L2-resident thereafter)
  prep_wt<<<cdiv(128 * 256, 256), 256, 0, stream>>>(W1, Wt1, 256, 128, 128);
  prep_wt<<<cdiv( 64 * 256, 256), 256, 0, stream>>>(W2, Wt2, 256,  64,  64);
  prep_wt<<<cdiv( 64 * 128, 256), 256, 0, stream>>>(W3, Wt3, 128,  64,  64);
  prep_wt<<<cdiv( 16 * 128, 256), 256, 0, stream>>>(W4, Wt4, 128,   8,  16);

  auto prologue = [&](const float* hin, int F) {
    long long n4 = (long long)N * F / 4;
    zero_f32x4<<<cdiv(n4, 256), 256, 0, stream>>>(agg, n4);
    long long work = (long long)E * (F >> 2);
    scatter_add_edges<<<cdiv(work, 256), 256, 0, stream>>>(hin, ei, ei + E, agg, E, F);
  };

  // Layer 1: 100k x 256 @ 256x128, ReLU+dropout
  prologue(x, 128);
  gemm_wmma_fused<256, 128, 128><<<N / 32, 256, 0, stream>>>(
      x, agg, Wt1, b1, bufA, 1, 0x13579BDFu);
  // Layer 2: 100k x 256 @ 256x64
  prologue(bufA, 128);
  gemm_wmma_fused<256, 64, 64><<<N / 32, 256, 0, stream>>>(
      bufA, agg, Wt2, b2, bufB, 1, 0x2468ACE0u);
  // Layer 3: 100k x 128 @ 128x64
  prologue(bufB, 64);
  gemm_wmma_fused<128, 64, 64><<<N / 32, 256, 0, stream>>>(
      bufB, agg, Wt3, b3, bufA, 1, 0x0F1E2D3Cu);
  // Layer 4: 100k x 128 @ 128x8 (padded to 16), no activation
  prologue(bufA, 64);
  gemm_wmma_fused<128, 16, 8><<<N / 32, 256, 0, stream>>>(
      bufA, agg, Wt4, b4, (float*)d_out, 0, 0u);
}
